// MultiHeadAttention_2379411882147
// MI455X (gfx1250) — matrix operand order repaired and checked by reading, not compiled, not run on hardware
//
#include <hip/hip_runtime.h>
#include <hip/hip_bf16.h>

// MI455X (gfx1250) fused causal MHA: f16 WMMA everywhere, flash-style
// attention, async global->LDS staging for K tiles, double-buffered LDS
// pipelines in the GEMMs. B=2, S=2048, D=1024, H=16, DK=64.

typedef __attribute__((ext_vector_type(16))) _Float16 v16h;
typedef __attribute__((ext_vector_type(8)))  _Float16 v8h;
typedef __attribute__((ext_vector_type(4)))  _Float16 v4h;
typedef __attribute__((ext_vector_type(8)))  float    v8f;

#define MHA_B  2
#define MHA_S  2048
#define MHA_D  1024
#define MHA_H  16
#define MHA_DK 64

// Assemble a v16h A/B fragment from two 16-byte LDS reads.
static __device__ inline v16h ld16(const _Float16* p0, const _Float16* p1) {
  v8h a = *(const v8h*)p0;
  v8h b = *(const v8h*)p1;
  v16h r;
#pragma unroll
  for (int i = 0; i < 8; ++i) { r[i] = a[i]; r[i + 8] = b[i]; }
  return r;
}

// Async copy 16 bytes global -> LDS (per lane). Tracked by ASYNCcnt.
// Operand order per CDNA5 ISA / VGLOBAL: vdst = LDS byte address, then the
// 64-bit global address VGPR pair, then saddr (off). Low 32 bits of a
// generic __shared__ pointer are the wave-relative LDS byte offset.
static __device__ inline void async_copy_b128(const void* gsrc, void* ldst) {
  unsigned long long ga = (unsigned long long)(uintptr_t)gsrc;
  unsigned int       la = (unsigned int)(uintptr_t)ldst;
  asm volatile("global_load_async_to_lds_b128 %0, %1, off"
               :: "v"(la), "v"(ga) : "memory");
}
static __device__ inline void wait_async0() {
  asm volatile("s_wait_asynccnt 0x0" ::: "memory");
}

// 64x64 f32 tile -> registers (8 x float4 per thread, 128 threads).
static __device__ inline void ld_tile_f32(const float* __restrict__ base,
                                          int row0, int k0, int t, float4* r) {
#pragma unroll
  for (int i = 0; i < 8; ++i) {
    int l  = t + 128 * i;
    int rr = l >> 4;
    int c4 = (l & 15) << 2;
    r[i] = *(const float4*)(base + (size_t)(row0 + rr) * MHA_D + k0 + c4);
  }
}
// registers -> 64x64 f16 LDS tile.
static __device__ inline void st_tile_f16(_Float16* lds, int t, const float4* r) {
#pragma unroll
  for (int i = 0; i < 8; ++i) {
    int l  = t + 128 * i;
    int rr = l >> 4;
    int c4 = (l & 15) << 2;
    v4h h = { (_Float16)r[i].x, (_Float16)r[i].y, (_Float16)r[i].z, (_Float16)r[i].w };
    *(v4h*)(lds + rr * 64 + c4) = h;
  }
}

// Batch of 4 WMMAs sharing an A fragment; B fragments are all loaded before
// the WMMA batch so the compiler can issue partial s_wait_dscnt waits.
static __device__ inline void wmma_row(v16h a, const _Float16* __restrict__ Bm,
                                       int l15, int hi, int ks, v8f* acc) {
  const _Float16* bp0 = Bm + (l15)      * 64 + ks * 32 + hi * 8;
  const _Float16* bp1 = Bm + (16 + l15) * 64 + ks * 32 + hi * 8;
  const _Float16* bp2 = Bm + (32 + l15) * 64 + ks * 32 + hi * 8;
  const _Float16* bp3 = Bm + (48 + l15) * 64 + ks * 32 + hi * 8;
  v16h b0 = ld16(bp0, bp0 + 16);
  v16h b1 = ld16(bp1, bp1 + 16);
  v16h b2 = ld16(bp2, bp2 + 16);
  v16h b3 = ld16(bp3, bp3 + 16);
  acc[0] = __builtin_amdgcn_wmma_f32_16x16x32_f16(false, a, false, b0, (short)0, acc[0], false, false);
  acc[1] = __builtin_amdgcn_wmma_f32_16x16x32_f16(false, a, false, b1, (short)0, acc[1], false, false);
  acc[2] = __builtin_amdgcn_wmma_f32_16x16x32_f16(false, a, false, b2, (short)0, acc[2], false, false);
  acc[3] = __builtin_amdgcn_wmma_f32_16x16x32_f16(false, a, false, b3, (short)0, acc[3], false, false);
}

// One 64-deep K-chunk of a 16x64 wave tile with A re-read from LDS.
static __device__ inline void wmma_chunk(const _Float16* __restrict__ A,
                                         const _Float16* __restrict__ Bm,
                                         int wave, int l15, int hi, v8f* acc) {
  const _Float16* arow = A + (wave * 16 + l15) * 64;
#pragma unroll
  for (int ks = 0; ks < 2; ++ks) {
    const _Float16* ap = arow + ks * 32 + hi * 8;
    v16h a = ld16(ap, ap + 16);
    wmma_row(a, Bm, l15, hi, ks, acc);
  }
}

// ---------------------------------------------------------------------------
// Projection GEMM: Out_headsplit[b][h][s][dk] = (X @ W^T)  (f32 in, f16 out)
// Grid: (BS/64, D/64), 128 threads = 4 waves; double-buffered LDS pipeline.
// ---------------------------------------------------------------------------
__global__ __launch_bounds__(128)
void mha_proj_gemm(const float* __restrict__ X,  // [BS, D]
                   const float* __restrict__ W,  // [D, D], row n = out feature
                   _Float16* __restrict__ Out)   // [B,H,S,DK] f16
{
  __shared__ _Float16 lA[2][64 * 64];
  __shared__ _Float16 lB[2][64 * 64];
  const int t = threadIdx.x;
  const int wave = t >> 5, lane = t & 31;
  const int l15 = lane & 15, hi = lane >> 4;
  const int m0 = blockIdx.x * 64;
  const int n0 = blockIdx.y * 64;

  v8f acc[4] = {};
  float4 ra[8], rb[8];
  ld_tile_f32(X, m0, 0, t, ra);
  ld_tile_f32(W, n0, 0, t, rb);

  for (int k0 = 0, it = 0; k0 < MHA_D; k0 += 64, ++it) {
    const int buf = it & 1;
    st_tile_f16(lA[buf], t, ra);
    st_tile_f16(lB[buf], t, rb);
    __syncthreads();
    if (k0 + 64 < MHA_D) {                 // prefetch next tile into registers
      ld_tile_f32(X, m0, k0 + 64, t, ra);
      ld_tile_f32(W, n0, k0 + 64, t, rb);
    }
    wmma_chunk(lA[buf], lB[buf], wave, l15, hi, acc);
  }

  // Store f16 head-split: n0 is a multiple of DK=64 -> one head per column.
  const int h = n0 / MHA_DK;
  const int mrow0 = m0 + wave * 16 + hi * 8;
#pragma unroll
  for (int c = 0; c < 4; ++c) {
    int dk = c * 16 + l15;
#pragma unroll
    for (int r = 0; r < 8; ++r) {
      int m = mrow0 + r;
      int b = m / MHA_S, s = m % MHA_S;
      Out[(((size_t)b * MHA_H + h) * MHA_S + s) * MHA_DK + dk] = (_Float16)acc[c][r];
    }
  }
}

// ---------------------------------------------------------------------------
// Flash attention (causal): per (qtile, h, b). 128 threads = 4 waves; each
// wave owns 16 query rows x DK=64 context. K tiles staged by async-to-LDS
// one tile ahead; V tiles staged through registers (transposed into LDS).
// Q A-fragments live in registers for the whole key loop.
// ---------------------------------------------------------------------------
__global__ __launch_bounds__(128)
void mha_flash_attn(const _Float16* __restrict__ Qh,
                    const _Float16* __restrict__ Kh,
                    const _Float16* __restrict__ Vh,
                    _Float16* __restrict__ Ctx)   // [B,S,D] f16 (heads merged)
{
  __shared__ _Float16 lQ[64 * 64];
  __shared__ _Float16 lK[2][64 * 64];     // [key][dk] row-major
  __shared__ _Float16 lVt[2][64 * 64];    // [dk][key] (transposed at store)
  __shared__ _Float16 lP[4][16 * 64];     // per-wave P tile for layout change

  const int t = threadIdx.x;
  const int wave = t >> 5, lane = t & 31;
  const int l15 = lane & 15, hi = lane >> 4;
  const int qb = blockIdx.x;
  const int h  = blockIdx.y;
  const int b  = blockIdx.z;

  const size_t head = (((size_t)b * MHA_H + h) * MHA_S) * MHA_DK;
  const _Float16* Qp = Qh + head;
  const _Float16* Kp = Kh + head;
  const _Float16* Vp = Vh + head;

  const int r8  = t >> 3;                 // 0..15 staging row helper
  const int c8  = (t & 7) << 3;           // 0..56 staging col helper

  // Stage Q tile (64x64 f16) once, synchronously.
#pragma unroll
  for (int i = 0; i < 4; ++i) {
    int rr = r8 + 16 * i;
    *(v8h*)(lQ + rr * 64 + c8) = *(const v8h*)(Qp + (size_t)(qb * 64 + rr) * MHA_DK + c8);
  }
  __syncthreads();

  // Q A-fragments (two k-steps of 32) held in registers for the whole loop.
  v16h qa[2];
  {
    const _Float16* ap0 = lQ + (wave * 16 + l15) * 64;
    qa[0] = ld16(ap0 + hi * 8,      ap0 + 16 + hi * 8);
    qa[1] = ld16(ap0 + 32 + hi * 8, ap0 + 48 + hi * 8);
  }

  v8f ctx[4] = {};
  float mrow[8], lrow[8];
#pragma unroll
  for (int r = 0; r < 8; ++r) { mrow[r] = -__builtin_inff(); lrow[r] = 0.0f; }

  const float scale = 0.125f;             // 1/sqrt(DK)
  const int nkt = qb + 1;                 // causal: key tiles 0..qb

  // Prologue: stage tile 0 (K async, V into registers).
  v8h vreg[4];
#pragma unroll
  for (int i = 0; i < 4; ++i) {
    int rr = r8 + 16 * i;
    async_copy_b128(Kp + (size_t)rr * MHA_DK + c8, lK[0] + rr * 64 + c8);
    vreg[i] = *(const v8h*)(Vp + (size_t)rr * MHA_DK + c8);
  }

  for (int kt = 0; kt < nkt; ++kt) {
    const int buf = kt & 1;

    // Transpose-store this tile's V registers into LDS.
#pragma unroll
    for (int i = 0; i < 4; ++i) {
      int rr = r8 + 16 * i;
#pragma unroll
      for (int j = 0; j < 8; ++j) lVt[buf][(c8 + j) * 64 + rr] = vreg[i][j];
    }
    wait_async0();                        // K tile kt fully in lK[buf]
    __syncthreads();

    // Stage next tile (K async into the other buffer, V into registers).
    if (kt + 1 < nkt) {
#pragma unroll
      for (int i = 0; i < 4; ++i) {
        int rr = r8 + 16 * i;
        async_copy_b128(Kp + (size_t)((kt + 1) * 64 + rr) * MHA_DK + c8,
                        lK[buf ^ 1] + rr * 64 + c8);
        vreg[i] = *(const v8h*)(Vp + (size_t)((kt + 1) * 64 + rr) * MHA_DK + c8);
      }
    }

    // S = Q K^T : B[k=dk][n=key] = K[key][dk] -> contiguous row reads of lK.
    // A fragments come straight from the register-resident qa[].
    v8f sc[4] = {};
    wmma_row(qa[0], lK[buf], l15, hi, 0, sc);
    wmma_row(qa[1], lK[buf], l15, hi, 1, sc);

    // Causal mask + scale + online softmax. C layout: row = r + 8*hi, col = l15.
#pragma unroll
    for (int r = 0; r < 8; ++r) {
      const int qrow = qb * 64 + wave * 16 + hi * 8 + r;
      float lm = -__builtin_inff();
#pragma unroll
      for (int c = 0; c < 4; ++c) {
        int kcol = kt * 64 + c * 16 + l15;
        float sv = sc[c][r] * scale;
        sv = (kcol <= qrow) ? sv : -__builtin_inff();
        sc[c][r] = sv;
        lm = fmaxf(lm, sv);
      }
#pragma unroll
      for (int msk = 1; msk < 16; msk <<= 1)
        lm = fmaxf(lm, __shfl_xor(lm, msk, 32));
      const float mnew = fmaxf(mrow[r], lm);
      const float corr = __expf(mrow[r] - mnew);
      float rsum = 0.0f;
#pragma unroll
      for (int c = 0; c < 4; ++c) {
        float p = __expf(sc[c][r] - mnew);
        sc[c][r] = p;
        rsum += p;
      }
#pragma unroll
      for (int msk = 1; msk < 16; msk <<= 1)
        rsum += __shfl_xor(rsum, msk, 32);
      lrow[r] = lrow[r] * corr + rsum;
      mrow[r] = mnew;
      ctx[0][r] *= corr; ctx[1][r] *= corr; ctx[2][r] *= corr; ctx[3][r] *= corr;
    }

    // P -> LDS (per-wave slab) to re-shape C-layout into A-layout.
    _Float16* Pp = lP[wave];
#pragma unroll
    for (int c = 0; c < 4; ++c)
#pragma unroll
      for (int r = 0; r < 8; ++r)
        Pp[(r + 8 * hi) * 64 + c * 16 + l15] = (_Float16)sc[c][r];

    // ctx += P (16x64) @ V (64x64): B[k=key][n=dk] = lVt[dk][key] row reads.
#pragma unroll
    for (int ks = 0; ks < 2; ++ks) {
      const _Float16* ap = Pp + l15 * 64 + ks * 32;
      v16h pa = ld16(ap + hi * 8, ap + 16 + hi * 8);
      wmma_row(pa, lVt[buf], l15, hi, ks, ctx);
    }
  }

  // Normalize and write merged-head context [B,S,D] in f16.
#pragma unroll
  for (int c = 0; c < 4; ++c) {
    int dk = c * 16 + l15;
#pragma unroll
    for (int r = 0; r < 8; ++r) {
      int s = qb * 64 + wave * 16 + hi * 8 + r;
      float v = ctx[c][r] / lrow[r];
      Ctx[((size_t)b * MHA_S + s) * MHA_D + h * MHA_DK + dk] = (_Float16)v;
    }
  }
}

// ---------------------------------------------------------------------------
// Output GEMM: out = ctx @ W_o^T + b_o  (f16 A, f32 W staged to f16, f32 out)
// ---------------------------------------------------------------------------
__global__ __launch_bounds__(128)
void mha_out_gemm(const _Float16* __restrict__ Xh, // [BS, D] f16
                  const float* __restrict__ W,     // [D, D]
                  const float* __restrict__ bias,  // [D]
                  float* __restrict__ Out)         // [BS, D] f32
{
  __shared__ _Float16 lA[2][64 * 64];
  __shared__ _Float16 lB[2][64 * 64];
  const int t = threadIdx.x;
  const int wave = t >> 5, lane = t & 31;
  const int l15 = lane & 15, hi = lane >> 4;
  const int m0 = blockIdx.x * 64;
  const int n0 = blockIdx.y * 64;
  const int r8 = t >> 3;
  const int c8 = (t & 7) << 3;

  v8f acc[4] = {};
  v8h ra[4];
  float4 rb[8];
#pragma unroll
  for (int i = 0; i < 4; ++i) {
    int rr = r8 + 16 * i;
    ra[i] = *(const v8h*)(Xh + (size_t)(m0 + rr) * MHA_D + c8);
  }
  ld_tile_f32(W, n0, 0, t, rb);

  for (int k0 = 0, it = 0; k0 < MHA_D; k0 += 64, ++it) {
    const int buf = it & 1;
#pragma unroll
    for (int i = 0; i < 4; ++i) {
      int rr = r8 + 16 * i;
      *(v8h*)(lA[buf] + rr * 64 + c8) = ra[i];
    }
    st_tile_f16(lB[buf], t, rb);
    __syncthreads();
    if (k0 + 64 < MHA_D) {
#pragma unroll
      for (int i = 0; i < 4; ++i) {
        int rr = r8 + 16 * i;
        ra[i] = *(const v8h*)(Xh + (size_t)(m0 + rr) * MHA_D + k0 + 64 + c8);
      }
      ld_tile_f32(W, n0, k0 + 64, t, rb);
    }
    wmma_chunk(lA[buf], lB[buf], wave, l15, hi, acc);
  }

  const int mrow0 = m0 + wave * 16 + hi * 8;
#pragma unroll
  for (int c = 0; c < 4; ++c) {
    int n = n0 + c * 16 + l15;
    float bn = bias[n];
#pragma unroll
    for (int r = 0; r < 8; ++r) {
      int m = mrow0 + r;
      Out[(size_t)m * MHA_D + n] = acc[c][r] + bn;
    }
  }
}

// ---------------------------------------------------------------------------
extern "C" void kernel_launch(void* const* d_in, const int* in_sizes, int n_in,
                              void* d_out, int out_size, void* d_ws, size_t ws_size,
                              hipStream_t stream) {
  (void)in_sizes; (void)n_in; (void)out_size; (void)ws_size;
  const float* q  = (const float*)d_in[0];
  const float* k  = (const float*)d_in[1];
  const float* v  = (const float*)d_in[2];
  const float* Wq = (const float*)d_in[3];
  const float* Wk = (const float*)d_in[4];
  const float* Wv = (const float*)d_in[5];
  const float* Wo = (const float*)d_in[6];
  const float* bo = (const float*)d_in[7];
  float* out = (float*)d_out;

  // Workspace: 4 x [B*S*D] f16 buffers = 32 MB total.
  const size_t perBuf = (size_t)MHA_B * MHA_S * MHA_D;
  _Float16* qh  = (_Float16*)d_ws;
  _Float16* kh  = qh + perBuf;
  _Float16* vh  = kh + perBuf;
  _Float16* ctx = vh + perBuf;

  dim3 gemm_grid((MHA_B * MHA_S) / 64, MHA_D / 64);
  mha_proj_gemm<<<gemm_grid, 128, 0, stream>>>(q, Wq, qh);
  mha_proj_gemm<<<gemm_grid, 128, 0, stream>>>(k, Wk, kh);
  mha_proj_gemm<<<gemm_grid, 128, 0, stream>>>(v, Wv, vh);

  dim3 attn_grid(MHA_S / 64, MHA_H, MHA_B);
  mha_flash_attn<<<attn_grid, 128, 0, stream>>>(qh, kh, vh, ctx);

  mha_out_gemm<<<gemm_grid, 128, 0, stream>>>(ctx, Wo, bo, out);
}
